// TopKWindowAttention_87600152969562
// MI455X (gfx1250) — matrix operand order repaired and checked by reading, not compiled, not run on hardware
//
#include <hip/hip_runtime.h>

typedef __attribute__((ext_vector_type(16))) _Float16 v16h;
typedef __attribute__((ext_vector_type(8)))  float    v8f;

#define WINSZ 7
#define NWSIDE 16        // windows per side (112/7)
#define NW 256           // total windows
#define WS 49            // tokens per window
#define TOPK 8
#define DMODEL 256
#define DH 32
#define NH 8
#define HW 112
#define PIX (HW*HW)      // 12544
#define S_TOT 648        // TOPK*WS + NW
#define S_FINE 392       // TOPK*WS
#define S_PAD 672        // 21*32 (pad S to multiple of WMMA K=32)
#define EPS_Z 1e-6f
#define PADV 0x7FFFFFFF  // sentinel for padded s entries

__device__ __forceinline__ float elu1(float x) {
    // elu(x)+1 :  x+1 for x>0, exp(x) for x<=0
    return x > 0.f ? x + 1.f : __expf(x);
}

// ---------------------------------------------------------------------------
// Kernel 1: window means qm/km/vm, shape (b, NW, D).
// ---------------------------------------------------------------------------
__global__ __launch_bounds__(128)
void wmean_kernel(const float* __restrict__ Q, const float* __restrict__ K,
                  const float* __restrict__ V,
                  float* __restrict__ qm, float* __restrict__ km,
                  float* __restrict__ vm) {
    int blk = blockIdx.x;               // b * DMODEL * NWSIDE
    int bi  = blk / (DMODEL * NWSIDE);
    int rem = blk % (DMODEL * NWSIDE);
    int c   = rem / NWSIDE;
    int wy  = rem % NWSIDE;

    __shared__ float cq[HW], ck[HW], cv[HW];
    int x = threadIdx.x;
    if (x < HW) {
        int base = (bi*DMODEL + c)*PIX + (wy*WINSZ)*HW + x;
        float sq = 0.f, sk = 0.f, sv = 0.f;
#pragma unroll
        for (int dy = 0; dy < WINSZ; ++dy) {
            sq += Q[base + dy*HW];
            sk += K[base + dy*HW];
            sv += V[base + dy*HW];
        }
        cq[x] = sq; ck[x] = sk; cv[x] = sv;
    }
    __syncthreads();
    if (x < NWSIDE) {
        float sq = 0.f, sk = 0.f, sv = 0.f;
#pragma unroll
        for (int dx = 0; dx < WINSZ; ++dx) {
            sq += cq[x*WINSZ + dx];
            sk += ck[x*WINSZ + dx];
            sv += cv[x*WINSZ + dx];
        }
        int o = (bi*NW + wy*NWSIDE + x)*DMODEL + c;
        const float inv = 1.f / (float)WS;
        qm[o] = sq*inv; km[o] = sk*inv; vm[o] = sv*inv;
    }
}

// ---------------------------------------------------------------------------
// Kernel 2: sim = qm . km^T per (b, query window), then top-8 indices.
// ---------------------------------------------------------------------------
__global__ __launch_bounds__(256)
void topk_kernel(const float* __restrict__ qm, const float* __restrict__ km,
                 int* __restrict__ topk) {
    int bi = blockIdx.x / NW;
    int qw = blockIdx.x % NW;
    int t  = threadIdx.x;

    __shared__ float qrow[DMODEL];
    __shared__ float sval[NW];
    __shared__ float red_v[NW];
    __shared__ int   red_i[NW];

    qrow[t] = qm[(bi*NW + qw)*DMODEL + t];
    __syncthreads();

    float dot = 0.f;
    const float* krow = km + (bi*NW + t)*DMODEL;
    for (int c = 0; c < DMODEL; ++c) dot += qrow[c] * krow[c];
    sval[t] = dot;
    __syncthreads();

    bool taken = false;
    for (int r = 0; r < TOPK; ++r) {
        red_v[t] = taken ? -3.4e38f : sval[t];
        red_i[t] = t;
        __syncthreads();
        for (int step = 128; step > 0; step >>= 1) {
            if (t < step) {
                float v2 = red_v[t + step]; int i2 = red_i[t + step];
                if (v2 > red_v[t] || (v2 == red_v[t] && i2 < red_i[t])) {
                    red_v[t] = v2; red_i[t] = i2;
                }
            }
            __syncthreads();
        }
        int win = red_i[0];
        if (t == 0) topk[(bi*NW + qw)*TOPK + r] = win;
        if (t == win) taken = true;
        __syncthreads();
    }
}

// helper: the 4 KV-tile WMMAs
#define WMMA4(a0,a1,b0,b1)                                                                            \
    acc00 = __builtin_amdgcn_wmma_f32_16x16x32_f16(false, a0, false, b0, (short)0, acc00, false, false); \
    acc01 = __builtin_amdgcn_wmma_f32_16x16x32_f16(false, a0, false, b1, (short)0, acc01, false, false); \
    acc10 = __builtin_amdgcn_wmma_f32_16x16x32_f16(false, a1, false, b0, (short)0, acc10, false, false); \
    acc11 = __builtin_amdgcn_wmma_f32_16x16x32_f16(false, a1, false, b1, (short)0, acc11, false, false);

// ---------------------------------------------------------------------------
// Kernel 3: linear attention per (b, window). 8 waves = 8 heads.
//   stage 1: KV(32x32) = Kf^T(32 x S) * Vn(S x 32), S streamed in 32-chunks
//            (12 branch-free fine chunks, 2 mixed chunks, 7 branch-free
//             coarse chunks with pure-arithmetic coalesced addressing)
//   stage 2: msg(49x32) = Qf(49x32) * KV(32x32), Z scaling applied per row
// ---------------------------------------------------------------------------
__global__ __launch_bounds__(256)
void attn_kernel(const float* __restrict__ Qp, const float* __restrict__ Kp,
                 const float* __restrict__ Vp, const float* __restrict__ km,
                 const float* __restrict__ vm, const int* __restrict__ topk,
                 float* __restrict__ outp) {
    const int bi   = blockIdx.x / NW;
    const int w    = blockIdx.x % NW;
    const int head = threadIdx.x >> 5;   // wave id == head (wave32)
    const int lane = threadIdx.x & 31;
    const int m    = lane & 15;
    const int grp  = lane >> 4;

    __shared__ int   s_idx[TOPK];
    __shared__ int   s_tab[S_PAD];         // >=0: fine pixel offset; <0: -(w2*D)-1; PADV: pad
    __shared__ int   s_qpix[64];           // pixel offset of query token l (0 for pad)
    __shared__ float kv_lds[NH][32][33];   // +1 pad vs LDS banks
    __shared__ float ksum_lds[NH][32];
    __shared__ float z_lds[NH][64];

    const int wy0 = (w >> 4)*WINSZ, wx0 = (w & 15)*WINSZ;

    if (threadIdx.x < TOPK)
        s_idx[threadIdx.x] = topk[(bi*NW + w)*TOPK + threadIdx.x];
    if (threadIdx.x < 64) {
        int l = threadIdx.x;
        s_qpix[l] = (l < WS) ? (wy0 + l/WINSZ)*HW + (wx0 + l%WINSZ) : 0;
    }
    __syncthreads();

    // build the s -> offset schedule (block-uniform, done once)
    for (int s = threadIdx.x; s < S_PAD; s += 256) {
        int val;
        if (s < S_FINE) {
            int j  = s / WS, e = s - j*WS;
            int w2 = s_idx[j];
            int y  = (w2 >> 4)*WINSZ + e/WINSZ;
            int x  = (w2 & 15)*WINSZ + (e % WINSZ);
            val = y*HW + x;
        } else if (s < S_TOT) {
            val = -((s - S_FINE)*DMODEL) - 1;
        } else {
            val = PADV;
        }
        s_tab[s] = val;
    }
    __syncthreads();

    // per-lane constant bases (A and B fragments share channels: head*DH + m)
    const int   c0    = head*DH + m;
    const int   baseF = (bi*DMODEL + c0)*PIX;        // fine base (Kp/Vp), ch c0
    const int   baseM = bi*NW*DMODEL + c0;           // mean base (km/vm), ch c0
    const float invS  = 1.0f / (float)S_TOT;

    v8f acc00 = {}, acc01 = {}, acc10 = {}, acc11 = {};
    float ks0 = 0.f, ks1 = 0.f;          // Ksum partials (d=m, d=m+16)

    // generic (mixed fine/coarse/pad) chunk, branchy path — used twice
    auto generic_chunk = [&](int s0) {
        v16h a0, a1, b0, b1;
#pragma unroll
        for (int hI = 0; hI < 16; ++hI) {
            const int kA = (hI < 8 ? hI : hI + 8) + 8*grp;
            const int kB = (hI < 8 ? hI : hI + 8) + 8*grp;
            float fa0 = 0.f, fa1 = 0.f, fb0 = 0.f, fb1 = 0.f;
            const int oA = s_tab[s0 + kA];
            if (oA != PADV) {
                float r0, r1;
                if (oA >= 0) { r0 = Kp[baseF + oA]; r1 = Kp[baseF + 16*PIX + oA]; }
                else { int mo = -1 - oA; r0 = km[baseM + mo]; r1 = km[baseM + 16 + mo]; }
                fa0 = elu1(r0); fa1 = elu1(r1);
            }
            const int oB = s_tab[s0 + kB];
            if (oB != PADV) {
                float r0, r1;
                if (oB >= 0) { r0 = Vp[baseF + oB]; r1 = Vp[baseF + 16*PIX + oB]; }
                else { int mo = -1 - oB; r0 = vm[baseM + mo]; r1 = vm[baseM + 16 + mo]; }
                fb0 = r0 * invS; fb1 = r1 * invS;
            }
            ks0 += fa0; ks1 += fa1;
            a0[hI] = (_Float16)fa0; a1[hI] = (_Float16)fa1;
            b0[hI] = (_Float16)fb0; b1[hI] = (_Float16)fb1;
        }
        WMMA4(a0, a1, b0, b1)
    };

    // ---- 12 pure-fine chunks (s in [0, 384), all offsets >= 0) ----
    for (int t = 0; t < 12; ++t) {
        const int s0 = t*32;

        // lookahead prefetch of next chunk's gathered lines
        {
            int on = s_tab[s0 + 32 + lane];
            if (on >= 0) {
                __builtin_prefetch(&Kp[baseF + on], 0, 0);
                __builtin_prefetch(&Vp[baseF + on], 0, 0);
            }
        }

        // wide LDS reads of this chunk's offsets (2x8-int runs for A and B)
        int offA[16], offB[16];
        {
            const int4 ta0 = *reinterpret_cast<const int4*>(&s_tab[s0 + 8*grp]);
            const int4 ta1 = *reinterpret_cast<const int4*>(&s_tab[s0 + 8*grp + 4]);
            const int4 ta2 = *reinterpret_cast<const int4*>(&s_tab[s0 + 16 + 8*grp]);
            const int4 ta3 = *reinterpret_cast<const int4*>(&s_tab[s0 + 16 + 8*grp + 4]);
            const int4 tb0 = *reinterpret_cast<const int4*>(&s_tab[s0 + 8*grp]);
            const int4 tb1 = *reinterpret_cast<const int4*>(&s_tab[s0 + 8*grp + 4]);
            const int4 tb2 = *reinterpret_cast<const int4*>(&s_tab[s0 + 16 + 8*grp]);
            const int4 tb3 = *reinterpret_cast<const int4*>(&s_tab[s0 + 16 + 8*grp + 4]);
            offA[0]=ta0.x; offA[1]=ta0.y; offA[2]=ta0.z;  offA[3]=ta0.w;
            offA[4]=ta1.x; offA[5]=ta1.y; offA[6]=ta1.z;  offA[7]=ta1.w;
            offA[8]=ta2.x; offA[9]=ta2.y; offA[10]=ta2.z; offA[11]=ta2.w;
            offA[12]=ta3.x;offA[13]=ta3.y;offA[14]=ta3.z; offA[15]=ta3.w;
            offB[0]=tb0.x; offB[1]=tb0.y; offB[2]=tb0.z;  offB[3]=tb0.w;
            offB[4]=tb1.x; offB[5]=tb1.y; offB[6]=tb1.z;  offB[7]=tb1.w;
            offB[8]=tb2.x; offB[9]=tb2.y; offB[10]=tb2.z; offB[11]=tb2.w;
            offB[12]=tb3.x;offB[13]=tb3.y;offB[14]=tb3.z; offB[15]=tb3.w;
        }

        v16h a0, a1, b0, b1;
#pragma unroll
        for (int hI = 0; hI < 16; ++hI) {
            const int oA = offA[hI];
            const int oB = offB[hI];
            float fa0 = elu1(Kp[baseF + oA]);
            float fa1 = elu1(Kp[baseF + 16*PIX + oA]);
            float fb0 = Vp[baseF + oB] * invS;
            float fb1 = Vp[baseF + 16*PIX + oB] * invS;
            ks0 += fa0; ks1 += fa1;
            a0[hI] = (_Float16)fa0; a1[hI] = (_Float16)fa1;
            b0[hI] = (_Float16)fb0; b1[hI] = (_Float16)fb1;
        }
        WMMA4(a0, a1, b0, b1)
    }

    // ---- mixed chunk: s in [384, 416) = 8 fine + 24 coarse ----
    generic_chunk(384);

    // ---- 7 pure-coarse chunks: s in [416, 640), offsets are arithmetic ----
    for (int t = 0; t < 7; ++t) {
        const int s0 = 416 + t*32;
        const int rowbase = (s0 - S_FINE)*DMODEL;   // uniform per chunk
        v16h a0, a1, b0, b1;
#pragma unroll
        for (int hI = 0; hI < 16; ++hI) {
            const int kA = (hI < 8 ? hI : hI + 8) + 8*grp;
            const int kB = (hI < 8 ? hI : hI + 8) + 8*grp;
            // coalesced: consecutive lanes hit consecutive channels
            float fa0 = elu1(km[baseM + rowbase + kA*DMODEL]);
            float fa1 = elu1(km[baseM + 16 + rowbase + kA*DMODEL]);
            float fb0 = vm[baseM + rowbase + kB*DMODEL] * invS;
            float fb1 = vm[baseM + 16 + rowbase + kB*DMODEL] * invS;
            ks0 += fa0; ks1 += fa1;
            a0[hI] = (_Float16)fa0; a1[hI] = (_Float16)fa1;
            b0[hI] = (_Float16)fb0; b1[hI] = (_Float16)fb1;
        }
        WMMA4(a0, a1, b0, b1)
    }

    // ---- final mixed chunk: s in [640, 672) = 8 coarse + 24 pad ----
    generic_chunk(640);

    // combine Ksum halves across the two 16-lane groups (wave32 xor-16)
    ks0 += __shfl_xor(ks0, 16, 32);
    ks1 += __shfl_xor(ks1, 16, 32);
    if (grp == 0) {
        ksum_lds[head][m]      = ks0;
        ksum_lds[head][m + 16] = ks1;
    }
    // spill KV tiles to LDS (D layout: lane n = lane&15, vgpr r -> m = r+8*grp)
#pragma unroll
    for (int r = 0; r < 8; ++r) {
        int mm = r + 8*grp;
        kv_lds[head][mm][m]           = acc00[r];
        kv_lds[head][mm][m + 16]      = acc01[r];
        kv_lds[head][mm + 16][m]      = acc10[r];
        kv_lds[head][mm + 16][m + 16] = acc11[r];
    }
    __syncthreads();

    // ---------------- stage 2: msg = Qf * KV, scaled by Z ----------------
    v16h bk0, bk1;
#pragma unroll
    for (int hI = 0; hI < 16; ++hI) {
        int kB = (hI < 8 ? hI : hI + 8) + 8*grp;
        bk0[hI] = (_Float16)kv_lds[head][kB][m];
        bk1[hI] = (_Float16)kv_lds[head][kB][m + 16];
    }

    const int qbase = (bi*DMODEL + head*DH)*PIX;
    for (int l0 = 0; l0 < 64; l0 += 16) {      // 4 query tiles (49 padded)
        const int lq   = l0 + m;
        const int qpix = qbase + s_qpix[lq];
        const bool qok = (lq < WS);
        v16h aq;
        float zp = 0.f;
#pragma unroll
        for (int hI = 0; hI < 16; ++hI) {
            int kA = (hI < 8 ? hI : hI + 8) + 8*grp;
            float f = 0.f;
            if (qok) f = elu1(Qp[qpix + kA*PIX]);
            zp += f * ksum_lds[head][kA];
            aq[hI] = (_Float16)f;
        }
        zp += __shfl_xor(zp, 16, 32);
        if (grp == 0) z_lds[head][lq] = (float)S_TOT / (zp + EPS_Z);
        __syncthreads();

        v8f zero0 = {}, zero1 = {};
        v8f o0 = __builtin_amdgcn_wmma_f32_16x16x32_f16(false, aq, false, bk0, (short)0, zero0, false, false);
        v8f o1 = __builtin_amdgcn_wmma_f32_16x16x32_f16(false, aq, false, bk1, (short)0, zero1, false, false);

#pragma unroll
        for (int r = 0; r < 8; ++r) {
            int l = l0 + r + 8*grp;
            if (l < WS) {
                float z = z_lds[head][l];
                int p = (bi*DMODEL + head*DH + m)*PIX + s_qpix[l];
                outp[p]          = o0[r] * z;
                outp[p + 16*PIX] = o1[r] * z;
            }
        }
    }
}

// ---------------------------------------------------------------------------
extern "C" void kernel_launch(void* const* d_in, const int* in_sizes, int n_in,
                              void* d_out, int out_size, void* d_ws, size_t ws_size,
                              hipStream_t stream) {
    (void)in_sizes; (void)out_size; (void)ws_size;
    if (n_in < 3) return;
    const float* q = (const float*)d_in[0];
    const float* k = (const float*)d_in[1];
    const float* v = (const float*)d_in[2];
    float* outp = (float*)d_out;

    // workspace layout: qm | km | vm | topk-indices
    float* qm   = (float*)d_ws;
    float* km   = qm + 2*NW*DMODEL;
    float* vm   = km + 2*NW*DMODEL;
    int*   topk = (int*)(vm + 2*NW*DMODEL);

    wmean_kernel<<<2*DMODEL*NWSIDE, 128, 0, stream>>>(q, k, v, qm, km, vm);
    topk_kernel <<<2*NW, 256, 0, stream>>>(qm, km, topk);
    attn_kernel <<<2*NW, 256, 0, stream>>>(q, k, v, km, vm, topk, outp);
}
